// StandardAttention_8143257993985
// MI455X (gfx1250) — matrix operand order repaired and checked by reading, not compiled, not run on hardware
//
#include <hip/hip_runtime.h>
#include <hip/hip_bf16.h>

// ---------------------------------------------------------------------------
// Problem constants (from reference): B=2, S=2048, D=2048, H=16, hd=128
// ---------------------------------------------------------------------------
#define BB   2
#define SS   2048
#define DD   2048
#define HH   16
#define HD   128

typedef __bf16 bf16_t;
typedef __attribute__((ext_vector_type(16))) __bf16 v16bf;
typedef __attribute__((ext_vector_type(8)))  __bf16 v8bf;
typedef __attribute__((ext_vector_type(8)))  float  v8f;

__device__ inline v8f zero8() {
    v8f r;
#pragma unroll
    for (int i = 0; i < 8; ++i) r[i] = 0.0f;
    return r;
}

__device__ inline v16bf combine(v8bf lo, v8bf hi) {
    v16bf r;
#pragma unroll
    for (int i = 0; i < 8; ++i) { r[i] = lo[i]; r[i + 8] = hi[i]; }
    return r;
}

// Reduce across the 16-lane half-groups of a wave32 (B/C/D matrix lane dim).
__device__ inline float rowmax16(float v) {
    v = fmaxf(v, __shfl_xor(v, 1, 32));
    v = fmaxf(v, __shfl_xor(v, 2, 32));
    v = fmaxf(v, __shfl_xor(v, 4, 32));
    v = fmaxf(v, __shfl_xor(v, 8, 32));
    return v;
}
__device__ inline float rowsum16(float v) {
    v += __shfl_xor(v, 1, 32);
    v += __shfl_xor(v, 2, 32);
    v += __shfl_xor(v, 4, 32);
    v += __shfl_xor(v, 8, 32);
    return v;
}

// ---------------------------------------------------------------------------
// GEMM fragment bundle: 2 A-subtiles (32 rows) x 4 B-subtiles (64 cols),
// double-buffered across K-chunks so loads of chunk k+32 overlap the 8 WMMAs
// of chunk k (LOADcnt decrements in order -> partial waits suffice).
// ---------------------------------------------------------------------------
struct Frags {
    v16bf a0, a1, b0, b1, b2, b3;
};

__device__ inline Frags load_frags(const bf16_t* arow0, const bf16_t* arow1,
                                   const bf16_t* br0, const bf16_t* br1,
                                   const bf16_t* br2, const bf16_t* br3,
                                   int k0, int k8, int k16) {
    Frags f;
    f.a0 = combine(*(const v8bf*)(arow0 + k0 + k16),
                   *(const v8bf*)(arow0 + k0 + 8 + k16));
    f.a1 = combine(*(const v8bf*)(arow1 + k0 + k16),
                   *(const v8bf*)(arow1 + k0 + 8 + k16));
    f.b0 = *(const v16bf*)(br0 + k0 + k16);
    f.b1 = *(const v16bf*)(br1 + k0 + k16);
    f.b2 = *(const v16bf*)(br2 + k0 + k16);
    f.b3 = *(const v16bf*)(br3 + k0 + k16);
    return f;
}

__device__ inline void wmma8(const Frags& f, v8f acc[2][4]) {
    acc[0][0] = __builtin_amdgcn_wmma_f32_16x16x32_bf16(
        false, f.a0, false, f.b0, (short)0, acc[0][0], false, false);
    acc[1][0] = __builtin_amdgcn_wmma_f32_16x16x32_bf16(
        false, f.a1, false, f.b0, (short)0, acc[1][0], false, false);
    acc[0][1] = __builtin_amdgcn_wmma_f32_16x16x32_bf16(
        false, f.a0, false, f.b1, (short)0, acc[0][1], false, false);
    acc[1][1] = __builtin_amdgcn_wmma_f32_16x16x32_bf16(
        false, f.a1, false, f.b1, (short)0, acc[1][1], false, false);
    acc[0][2] = __builtin_amdgcn_wmma_f32_16x16x32_bf16(
        false, f.a0, false, f.b2, (short)0, acc[0][2], false, false);
    acc[1][2] = __builtin_amdgcn_wmma_f32_16x16x32_bf16(
        false, f.a1, false, f.b2, (short)0, acc[1][2], false, false);
    acc[0][3] = __builtin_amdgcn_wmma_f32_16x16x32_bf16(
        false, f.a0, false, f.b3, (short)0, acc[0][3], false, false);
    acc[1][3] = __builtin_amdgcn_wmma_f32_16x16x32_bf16(
        false, f.a1, false, f.b3, (short)0, acc[1][3], false, false);
}

// ---------------------------------------------------------------------------
// Kernel 1: fp32 -> bf16 conversion, 8 elements/thread (b128 in / b128 out)
// ---------------------------------------------------------------------------
__global__ void cvt_bf16_kernel(const float* __restrict__ src,
                                bf16_t* __restrict__ dst, int n8) {
    int i = blockIdx.x * blockDim.x + threadIdx.x;
    if (i < n8) {
        v8f v = *(const v8f*)(src + (size_t)i * 8);
        v8bf o;
#pragma unroll
        for (int e = 0; e < 8; ++e) o[e] = (bf16_t)v[e];
        *(v8bf*)(dst + (size_t)i * 8) = o;
    }
}

// ---------------------------------------------------------------------------
// Kernel 2: Q/K/V projection.  C[row, f] = sum_d x[row, d] * W[f, d]
// One wave -> 32x64 output region; double-buffered K-loop.
// Q, K written [b][h][s][hd]; V written transposed [b][h][hd][s].
// grid = (128 row tiles, 4 col groups, 3 matrices), block = 256 (8 waves)
// ---------------------------------------------------------------------------
__global__ void __launch_bounds__(256)
proj_qkv_kernel(const bf16_t* __restrict__ xb,
                const bf16_t* __restrict__ Wq,
                const bf16_t* __restrict__ Wk,
                const bf16_t* __restrict__ Wv,
                bf16_t* __restrict__ Q,
                bf16_t* __restrict__ K,
                bf16_t* __restrict__ Vt) {
    const int wave = threadIdx.x >> 5;
    const int tid  = threadIdx.x & 31;
    const int n    = tid & 15;       // column within tile (B/C/D lane dim)
    const int hi   = tid >> 4;       // which 16-lane half
    const int k8   = hi * 8;         // unused; A and B fragments both use k16
    const int k16  = hi * 16;        // B-fragment K sub-offset
    const int mb   = hi * 8;         // C-fragment row offset
    const int am   = tid & 15;       // A-matrix row per lane

    const int row0 = blockIdx.x * 32;
    const int col0 = (blockIdx.y * 8 + wave) * 64;
    const int mat  = blockIdx.z;
    const bf16_t* W = (mat == 0) ? Wq : ((mat == 1) ? Wk : Wv);

    const bf16_t* arow0 = xb + (size_t)(row0 + am) * DD;
    const bf16_t* arow1 = arow0 + (size_t)16 * DD;
    const bf16_t* br0 = W + (size_t)(col0 +  0 + n) * DD;
    const bf16_t* br1 = W + (size_t)(col0 + 16 + n) * DD;
    const bf16_t* br2 = W + (size_t)(col0 + 32 + n) * DD;
    const bf16_t* br3 = W + (size_t)(col0 + 48 + n) * DD;

    v8f acc[2][4];
#pragma unroll
    for (int mi = 0; mi < 2; ++mi)
#pragma unroll
        for (int j = 0; j < 4; ++j) acc[mi][j] = zero8();

    Frags cur = load_frags(arow0, arow1, br0, br1, br2, br3, 0, k8, k16);
    for (int k0 = 32; k0 < DD; k0 += 32) {
        Frags nxt = load_frags(arow0, arow1, br0, br1, br2, br3, k0, k8, k16);
        wmma8(cur, acc);
        cur = nxt;
    }
    wmma8(cur, acc);

#pragma unroll
    for (int mi = 0; mi < 2; ++mi)
#pragma unroll
        for (int j = 0; j < 4; ++j) {
            const int f = col0 + 16 * j + n;
            const int h = f >> 7;
            const int d = f & (HD - 1);
#pragma unroll
            for (int i = 0; i < 8; ++i) {
                const int row = row0 + mi * 16 + i + mb;   // global (b*S + s)
                const int b   = row >> 11;
                const int s   = row & (SS - 1);
                const bf16_t v = (bf16_t)acc[mi][j][i];
                if (mat == 0)
                    Q[(((size_t)b * HH + h) * SS + s) * HD + d] = v;
                else if (mat == 1)
                    K[(((size_t)b * HH + h) * SS + s) * HD + d] = v;
                else
                    Vt[(((size_t)b * HH + h) * HD + d) * SS + s] = v;
            }
        }
}

// ---------------------------------------------------------------------------
// Kernel 3: RoPE over Q and K (in place, [b][h][s][hd] layout)
// thread t -> (bh, s, i<64): pairs (i, i+64); cos/sin repeat over halves
// ---------------------------------------------------------------------------
__global__ void rope_kernel(bf16_t* __restrict__ Q, bf16_t* __restrict__ K,
                            const float* __restrict__ cosp,
                            const float* __restrict__ sinp) {
    const int t  = blockIdx.x * blockDim.x + threadIdx.x;
    const int i  = t & 63;
    const int s  = (t >> 6) & (SS - 1);
    const int bh = t >> 17;
    const size_t base = ((size_t)bh * SS + s) * HD;
    const float c  = cosp[s * HD + i];
    const float sn = sinp[s * HD + i];

    float q0 = (float)Q[base + i], q1 = (float)Q[base + i + 64];
    Q[base + i]      = (bf16_t)(q0 * c - q1 * sn);
    Q[base + i + 64] = (bf16_t)(q1 * c + q0 * sn);

    float k0 = (float)K[base + i], k1 = (float)K[base + i + 64];
    K[base + i]      = (bf16_t)(k0 * c - k1 * sn);
    K[base + i + 64] = (bf16_t)(k1 * c + k0 * sn);
}

// ---------------------------------------------------------------------------
// Kernel 4: flash attention (causal, online softmax, bf16 WMMA / fp32 acc)
// 8 waves per WG, each wave owns 16 query rows; per 32-key block:
//   8 WMMAs (Q*K^T over hd=128) + 8 WMMAs (P*V over 8 d-tiles of 16)
// P goes C-layout -> A-layout through a per-wave 1KB LDS tile.
// grid = (S/128, H, B), block = 256
// ---------------------------------------------------------------------------
__global__ void __launch_bounds__(256)
attn_kernel(const bf16_t* __restrict__ Q,
            const bf16_t* __restrict__ K,
            const bf16_t* __restrict__ Vt,
            bf16_t* __restrict__ ctx) {
    __shared__ __align__(16) bf16_t Plds[8 * 512];   // 16x32 bf16 per wave

    const int wave = threadIdx.x >> 5;
    const int tid  = threadIdx.x & 31;
    const int n    = tid & 15;
    const int hi   = tid >> 4;
    const int k8   = hi * 8;
    const int k16  = hi * 16;
    const int mb   = hi * 8;
    const int am   = tid & 15;

    const int h  = blockIdx.y;
    const int b  = blockIdx.z;
    const int bh = b * HH + h;
    const int qb = blockIdx.x * 128 + wave * 16;

    const bf16_t* Qh = Q  + ((size_t)bh * SS + qb) * HD;
    const bf16_t* Kh = K  + (size_t)bh * SS * HD;
    const bf16_t* Vh = Vt + (size_t)bh * HD * SS;
    bf16_t* lp = Plds + wave * 512;

    // Q fragment: 16 rows x 128 (4 K-chunks of 32), A-layout, kept resident
    v16bf aq[4];
#pragma unroll
    for (int c = 0; c < 4; ++c) {
        const bf16_t* r = Qh + (size_t)am * HD + c * 32;
        aq[c] = combine(*(const v8bf*)(r + k16), *(const v8bf*)(r + 8 + k16));
    }

    v8f O[8];
#pragma unroll
    for (int t = 0; t < 8; ++t) O[t] = zero8();
    float M[8], L[8];
#pragma unroll
    for (int i = 0; i < 8; ++i) { M[i] = -1e30f; L[i] = 0.0f; }

    const float scale = 0.08838834764831845f;   // 1/sqrt(128)
    const int nkb = (qb + 47) >> 5;             // causal 32-key blocks

    for (int j = 0; j < nkb; ++j) {
        const int kb = j << 5;

        // ---- logits: two 16-key n-subtiles, reduce over hd=128 ----
        v8f s0 = zero8(), s1 = zero8();
#pragma unroll
        for (int c = 0; c < 4; ++c) {
            v16bf b0 = *(const v16bf*)(Kh + (size_t)(kb + n)      * HD + c * 32 + k16);
            v16bf b1 = *(const v16bf*)(Kh + (size_t)(kb + 16 + n) * HD + c * 32 + k16);
            s0 = __builtin_amdgcn_wmma_f32_16x16x32_bf16(
                false, aq[c], false, b0, (short)0, s0, false, false);
            s1 = __builtin_amdgcn_wmma_f32_16x16x32_bf16(
                false, aq[c], false, b1, (short)0, s1, false, false);
        }

        // ---- online softmax update ----
        float p0[8], p1[8], alpha[8];
#pragma unroll
        for (int i = 0; i < 8; ++i) {
            const int grow = qb + i + mb;
            float v0 = s0[i] * scale;
            float v1 = s1[i] * scale;
            if (kb + n      > grow) v0 = -1e30f;   // causal mask
            if (kb + 16 + n > grow) v1 = -1e30f;
            float mn = rowmax16(fmaxf(v0, v1));
            mn = fmaxf(mn, M[i]);
            alpha[i] = __expf(M[i] - mn);
            p0[i] = __expf(v0 - mn);
            p1[i] = __expf(v1 - mn);
            float rs = rowsum16(p0[i] + p1[i]);
            L[i] = L[i] * alpha[i] + rs;
            M[i] = mn;
        }
#pragma unroll
        for (int t = 0; t < 8; ++t)
#pragma unroll
            for (int i = 0; i < 8; ++i) O[t][i] *= alpha[i];

        // ---- P: C-layout -> A-layout via per-wave LDS tile (same-wave DS
        //      ops are architecturally in-order; no barrier needed) ----
#pragma unroll
        for (int i = 0; i < 8; ++i) {
            lp[(i + mb) * 32 + n]      = (bf16_t)p0[i];
            lp[(i + mb) * 32 + 16 + n] = (bf16_t)p1[i];
        }
        v16bf ap = combine(*(const v8bf*)(lp + am * 32 + k16),
                           *(const v8bf*)(lp + am * 32 + 8 + k16));

        // ---- O += P * V (V transposed -> contiguous B fragments) ----
#pragma unroll
        for (int t = 0; t < 8; ++t) {
            v16bf bv = *(const v16bf*)(Vh + (size_t)(t * 16 + n) * SS + kb + k16);
            O[t] = __builtin_amdgcn_wmma_f32_16x16x32_bf16(
                false, ap, false, bv, (short)0, O[t], false, false);
        }
    }

    // ---- normalize and store ctx as [b][s][h*128 + d] (row-major D) ----
#pragma unroll
    for (int i = 0; i < 8; ++i) {
        const int grow = qb + i + mb;
        const float inv = 1.0f / L[i];
        bf16_t* dst = ctx + ((size_t)b * SS + grow) * DD + h * HD;
#pragma unroll
        for (int t = 0; t < 8; ++t)
            dst[t * 16 + n] = (bf16_t)(O[t][i] * inv);
    }
}

// ---------------------------------------------------------------------------
// Kernel 5: output projection  out[row, f] = sum_d ctx[row, d] * Wo[f, d]
// Same 2x4 register blocking + double buffering; fp32 output.
// grid = (128, 4, 1), block = 256
// ---------------------------------------------------------------------------
__global__ void __launch_bounds__(256)
out_proj_kernel(const bf16_t* __restrict__ ctx,
                const bf16_t* __restrict__ Wo,
                float* __restrict__ out) {
    const int wave = threadIdx.x >> 5;
    const int tid  = threadIdx.x & 31;
    const int n    = tid & 15;
    const int hi   = tid >> 4;
    const int k8   = hi * 8;
    const int k16  = hi * 16;
    const int mb   = hi * 8;
    const int am   = tid & 15;

    const int row0 = blockIdx.x * 32;
    const int col0 = (blockIdx.y * 8 + wave) * 64;

    const bf16_t* arow0 = ctx + (size_t)(row0 + am) * DD;
    const bf16_t* arow1 = arow0 + (size_t)16 * DD;
    const bf16_t* br0 = Wo + (size_t)(col0 +  0 + n) * DD;
    const bf16_t* br1 = Wo + (size_t)(col0 + 16 + n) * DD;
    const bf16_t* br2 = Wo + (size_t)(col0 + 32 + n) * DD;
    const bf16_t* br3 = Wo + (size_t)(col0 + 48 + n) * DD;

    v8f acc[2][4];
#pragma unroll
    for (int mi = 0; mi < 2; ++mi)
#pragma unroll
        for (int j = 0; j < 4; ++j) acc[mi][j] = zero8();

    Frags cur = load_frags(arow0, arow1, br0, br1, br2, br3, 0, k8, k16);
    for (int k0 = 32; k0 < DD; k0 += 32) {
        Frags nxt = load_frags(arow0, arow1, br0, br1, br2, br3, k0, k8, k16);
        wmma8(cur, acc);
        cur = nxt;
    }
    wmma8(cur, acc);

#pragma unroll
    for (int mi = 0; mi < 2; ++mi)
#pragma unroll
        for (int j = 0; j < 4; ++j)
#pragma unroll
            for (int i = 0; i < 8; ++i) {
                const int row = row0 + mi * 16 + i + mb;
                out[(size_t)row * DD + col0 + 16 * j + n] = acc[mi][j][i];
            }
}

// ---------------------------------------------------------------------------
// Launch
// inputs: x, cos, sin, Wq, Wk, Wv, Wo   (all fp32)
// ---------------------------------------------------------------------------
extern "C" void kernel_launch(void* const* d_in, const int* in_sizes, int n_in,
                              void* d_out, int out_size, void* d_ws, size_t ws_size,
                              hipStream_t stream) {
    (void)in_sizes; (void)n_in; (void)out_size; (void)ws_size;

    const float* x    = (const float*)d_in[0];
    const float* cosp = (const float*)d_in[1];
    const float* sinp = (const float*)d_in[2];
    const float* Wq   = (const float*)d_in[3];
    const float* Wk   = (const float*)d_in[4];
    const float* Wv   = (const float*)d_in[5];
    const float* Wo   = (const float*)d_in[6];
    float* out = (float*)d_out;

    const size_t XN = (size_t)BB * SS * DD;   // 8,388,608
    const size_t WN = (size_t)DD * DD;        // 4,194,304

    bf16_t* p    = (bf16_t*)d_ws;
    bf16_t* xb   = p; p += XN;
    bf16_t* wqb  = p; p += WN;
    bf16_t* wkb  = p; p += WN;
    bf16_t* wvb  = p; p += WN;
    bf16_t* wob  = p; p += WN;
    bf16_t* Qb   = p; p += XN;
    bf16_t* Kb   = p; p += XN;
    bf16_t* Vtb  = p; p += XN;
    bf16_t* ctxb = p; p += XN;

    // 1) convert fp32 -> bf16 (vectorized 8/thread)
    cvt_bf16_kernel<<<(int)(XN / 8 / 256), 256, 0, stream>>>(x,  xb,  (int)(XN / 8));
    cvt_bf16_kernel<<<(int)(WN / 8 / 256), 256, 0, stream>>>(Wq, wqb, (int)(WN / 8));
    cvt_bf16_kernel<<<(int)(WN / 8 / 256), 256, 0, stream>>>(Wk, wkb, (int)(WN / 8));
    cvt_bf16_kernel<<<(int)(WN / 8 / 256), 256, 0, stream>>>(Wv, wvb, (int)(WN / 8));
    cvt_bf16_kernel<<<(int)(WN / 8 / 256), 256, 0, stream>>>(Wo, wob, (int)(WN / 8));

    // 2) Q/K/V projections (double-buffered WMMA GEMM)
    proj_qkv_kernel<<<dim3(BB * SS / 32, DD / 64 / 8, 3), 256, 0, stream>>>(
        xb, wqb, wkb, wvb, Qb, Kb, Vtb);

    // 3) RoPE on Q and K
    rope_kernel<<<(int)((size_t)BB * HH * SS * 64 / 256), 256, 0, stream>>>(
        Qb, Kb, cosp, sinp);

    // 4) causal flash attention (WMMA)
    attn_kernel<<<dim3(SS / 128, HH, BB), 256, 0, stream>>>(Qb, Kb, Vtb, ctxb);

    // 5) output projection (double-buffered WMMA GEMM, fp32 out)
    out_proj_kernel<<<dim3(BB * SS / 32, DD / 64 / 8, 1), 256, 0, stream>>>(
        ctxb, wob, out);
}
